// Int8Linear_52072183497434
// MI455X (gfx1250) — matrix operand order repaired and checked by reading, not compiled, not run on hardware
//
#include <hip/hip_runtime.h>

// ---------------------------------------------------------------------------
// Int8 group-quantized linear:  out = x @ dequant(w, scale)^T + bias
//   x      : [4096, 4096]  f32
//   w      : [11008, 4096] int32 (values 0..126, exact in bf16)
//   scale  : [11008, 32]   f32   (one scale per 128-wide K group)
//   bias   : [11008]       f32
//   out    : [4096, 11008] f32
//
// Strategy: v_wmma_f32_16x16x32_bf16. Weights are exact in bf16; the group
// scale is factored OUT of the WMMA via per-group f32 accumulators, so the
// only rounding is x's f32->bf16 conversion.
// ---------------------------------------------------------------------------

typedef __attribute__((ext_vector_type(16))) __bf16 v16bf;
typedef __attribute__((ext_vector_type(8)))  __bf16 v8bf;
typedef __attribute__((ext_vector_type(8)))  float  v8f;

#define TOKENS   4096
#define IN_F     4096
#define OUT_F    11008
#define NGROUPS  32
#define GSIZE    128

// Block tile: 64 M x 128 N. 8 waves; each wave computes a 64M x 16N C strip
// (4 stacked 16x16 WMMA tiles) so the B-fragment int->bf16 conversion is
// amortized over 4 WMMAs.
#define BM 64
#define BN 128
// LDS x-tile row stride: 136 bf16 = 272 B = 17 * 16 B  (b128-aligned, rows
// offset by 4 banks -> conflict-free 16-lane b128 reads).
#define LDS_STRIDE8 17

__global__ __launch_bounds__(256) void int8linear_wmma(
    const float* __restrict__ x,
    const int*   __restrict__ w,
    const float* __restrict__ scale,
    const float* __restrict__ bias,
    float*       __restrict__ out)
{
    __shared__ v8bf xs[BM * LDS_STRIDE8];   // 64 x 136 bf16 = 17 KB

    const int tid  = threadIdx.x;
    const int wave = tid >> 5;
    const int lane = tid & 31;
    const int m    = lane & 15;   // A row / C col index within tile
    const int h    = lane >> 4;   // K-half selector (A/B fragment layout)

    const int mBase = blockIdx.y * BM;
    const int o     = blockIdx.x * BN + wave * 16 + m;   // output feature (B col / C col)

    v8f c[4];
    #pragma unroll
    for (int t = 0; t < 4; ++t) c[t] = v8f{0.f,0.f,0.f,0.f,0.f,0.f,0.f,0.f};

    const float* srow = scale + (size_t)o * NGROUPS;

    for (int kg = 0; kg < NGROUPS; ++kg) {
        __syncthreads();   // previous group's LDS reads done before overwrite

        // ---- Stage x tile [64 x 128] f32 -> bf16 into LDS (coalesced) ----
        #pragma unroll
        for (int r = 0; r < 4; ++r) {
            const int g   = tid + 256 * r;   // 0..1023 groups of 8 floats
            const int row = g >> 4;          // 0..63
            const int cg  = g & 15;          // 8-column group 0..15
            const float4* xg =
                (const float4*)(x + (size_t)(mBase + row) * IN_F + kg * GSIZE + cg * 8);
            const float4 f0 = xg[0];
            const float4 f1 = xg[1];
            const float f[8] = {f0.x, f0.y, f0.z, f0.w, f1.x, f1.y, f1.z, f1.w};
            v8bf b8;
            #pragma unroll
            for (int j = 0; j < 8; ++j) b8[j] = (__bf16)f[j];
            xs[row * LDS_STRIDE8 + cg] = b8;
        }
        __syncthreads();

        const float s = srow[kg];            // per-lane group scale (col = lane&15)

        v8f cg4[4];
        #pragma unroll
        for (int t = 0; t < 4; ++t) cg4[t] = v8f{0.f,0.f,0.f,0.f,0.f,0.f,0.f,0.f};

        #pragma unroll
        for (int kk = 0; kk < 4; ++kk) {
            // ---- B fragment: 16 consecutive K of w[o, :] (64 B contiguous) ----
            // B k order used here (contiguous): lanes 0-15 hold K=0..15, lanes 16-31 K=16..31.
            const int4* wp =
                (const int4*)(w + (size_t)o * IN_F + kg * GSIZE + kk * 32 + h * 16);
            int wv[16];
            *(int4*)(wv +  0) = wp[0];
            *(int4*)(wv +  4) = wp[1];
            *(int4*)(wv +  8) = wp[2];
            *(int4*)(wv + 12) = wp[3];
            v16bf B;
            #pragma unroll
            for (int j = 0; j < 16; ++j) B[j] = (__bf16)(float)wv[j];  // exact

            // ---- 4 A fragments (stacked M tiles), WMMA into group accum ----
            #pragma unroll
            for (int t = 0; t < 4; ++t) {
                // A k order used here (contiguous): lane m = row; elems 0..15 = K h*16+0..15
                // -> two 16-B LDS reads.
                const int base = (t * 16 + m) * LDS_STRIDE8 + kk * 4 + h * 2;
                const v8bf a0 = xs[base];
                const v8bf a1 = xs[base + 1];
                const v16bf A = __builtin_shufflevector(
                    a0, a1, 0,1,2,3,4,5,6,7,8,9,10,11,12,13,14,15);
                cg4[t] = __builtin_amdgcn_wmma_f32_16x16x32_bf16(
                    /*neg_a=*/false, A, /*neg_b=*/false, B,
                    /*c_mod=*/(short)0, cg4[t],
                    /*reuse_a=*/false, /*reuse_b=*/false);
            }
        }

        // Apply group scale once per 128-K group (weights entered WMMA exact).
        #pragma unroll
        for (int t = 0; t < 4; ++t) c[t] += cg4[t] * s;
    }

    // ---- Epilogue: bias + store. C layout: VGPR v -> row v + 8*h, col m ----
    const float bv = bias[o];
    #pragma unroll
    for (int t = 0; t < 4; ++t) {
        #pragma unroll
        for (int v = 0; v < 8; ++v) {
            const int row = mBase + t * 16 + v + 8 * h;
            out[(size_t)row * OUT_F + o] = c[t][v] + bv;
        }
    }
}

extern "C" void kernel_launch(void* const* d_in, const int* in_sizes, int n_in,
                              void* d_out, int out_size, void* d_ws, size_t ws_size,
                              hipStream_t stream) {
    (void)in_sizes; (void)n_in; (void)out_size; (void)d_ws; (void)ws_size;
    const float* x     = (const float*)d_in[0];
    const int*   w     = (const int*)  d_in[1];
    const float* scale = (const float*)d_in[2];
    const float* bias  = (const float*)d_in[3];
    float*       out   = (float*)d_out;

    dim3 grid(OUT_F / BN, TOKENS / BM);   // 86 x 64 blocks
    dim3 block(256);                      // 8 wave32 waves
    int8linear_wmma<<<grid, block, 0, stream>>>(x, w, scale, bias, out);
}
